// OpticalLinearLayer_25245817766446
// MI455X (gfx1250) — matrix operand order repaired and checked by reading, not compiled, not run on hardware
//
#include <hip/hip_runtime.h>

// Optical interferometer linear layer for MI455X (gfx1250, wave32).
//
// Phase 1 (build_T): T_eff rows evolve independently under column-Givens ops,
//   so 16 workgroups each carry 4 rows of the 64x256 complex matrix in LDS,
//   applying the 256 layers in reverse (right-multiplication order). Exact
//   fp32 math matching the reference's complex64 build, rounded to f16 once.
//
// Phase 2 (wmma_gemm): out = x @ T_eff^T.
//   168 MB HBM traffic @ 23.3 TB/s => ~7.2 us floor. fp32 WMMA (K=4/issue)
//   would be compute-bound; f16 WMMA (K=32/issue) with f32 accumulate keeps
//   this memory-bound. x stays fp32 in HBM (traffic unchanged), converted to
//   f16 while staging into LDS. T_eff (f16, 32 KB, L2-hot) is staged with the
//   CDNA5 async-to-LDS DMA path (ASYNCcnt), overlapping the x conversion.

#define NUM_MODES   256
#define IN_FEAT     256
#define OUT_FEAT    64
#define LAYERS      256
#define PAIRS       128

typedef __attribute__((ext_vector_type(16))) _Float16 v16h;
typedef __attribute__((ext_vector_type(8)))  _Float16 v8h;
typedef __attribute__((ext_vector_type(4)))  _Float16 v4h;
typedef __attribute__((ext_vector_type(8)))  float    v8f;
typedef __attribute__((ext_vector_type(4)))  float    v4f;

#define PI_F  3.14159265358979323846f
#define TWO_PI_F 6.28318530717958647692f

// ---------------------------------------------------------------------------
// Phase 1: build T_eff (64x256) in f16 into workspace.
// Each block owns 4 rows of E = Sel * D(out_phases) * L_256 * ... * L_1.
// Column update for layer l (applied from the right, so reverse layer order):
//   new col m1 = e^{i phi} (c*col_m1 + s*col_m2)
//   new col m2 = -s*col_m1 + c*col_m2
// ---------------------------------------------------------------------------
__global__ __launch_bounds__(256) void build_T(
    const float* __restrict__ theta, const float* __restrict__ phi,
    const float* __restrict__ opha,  const int* __restrict__ mode1,
    const int* __restrict__ mode2,   _Float16* __restrict__ Tf16)
{
  __shared__ float Er[4][NUM_MODES];
  __shared__ float Ei[4][NUM_MODES];
  __shared__ float Cs[PAIRS], Sn[PAIRS], Ph_r[PAIRS], Ph_i[PAIRS];
  __shared__ int   M1[PAIRS], M2[PAIRS];

  const int t  = threadIdx.x;
  const int r0 = blockIdx.x * 4;      // global output rows r0..r0+3

  // init: E[rr][j] = e^{i * opha[r0+rr]} * delta_{r0+rr, j}
  for (int idx = t; idx < 4 * NUM_MODES; idx += 256) {
    int rr = idx >> 8, j = idx & 255;
    int gi = r0 + rr;
    float vr = 0.0f, vi = 0.0f;
    if (j == gi) { vr = cosf(opha[gi]); vi = sinf(opha[gi]); }
    Er[rr][j] = vr; Ei[rr][j] = vi;
  }
  __syncthreads();

  for (int l = LAYERS - 1; l >= 0; --l) {
    if (t < PAIRS) {
      float th = theta[l * PAIRS + t];
      float ph = phi[l * PAIRS + t];
      th = fmodf(th, PI_F);                       // theta % pi
      if (th > 0.5f * PI_F) { th = PI_F - th; ph += PI_F; }  // flip branch
      ph = fmodf(ph, TWO_PI_F);
      Cs[t]  = cosf(th);  Sn[t]  = sinf(th);
      Ph_r[t] = cosf(ph); Ph_i[t] = sinf(ph);
      M1[t] = mode1[l * PAIRS + t];
      M2[t] = mode2[l * PAIRS + t];
    }
    __syncthreads();

    // 128 pairs x 4 rows = 512 disjoint column-pair updates
    for (int idx = t; idx < PAIRS * 4; idx += 256) {
      int p  = idx >> 2, rr = idx & 3;
      int m1 = M1[p], m2 = M2[p];
      float c  = Cs[p],   s  = Sn[p];
      float er = Ph_r[p], ei = Ph_i[p];
      float ur = Er[rr][m1], ui = Ei[rr][m1];
      float vr = Er[rr][m2], vi = Ei[rr][m2];
      float tr = c * ur + s * vr;
      float ti = c * ui + s * vi;
      Er[rr][m1] = er * tr - ei * ti;
      Ei[rr][m1] = er * ti + ei * tr;
      Er[rr][m2] = c * vr - s * ur;
      Ei[rr][m2] = c * vi - s * ui;
    }
    __syncthreads();
  }

  // T_eff[i][j] = Re(E[i][j]); single rounding to f16
  for (int idx = t; idx < 4 * NUM_MODES; idx += 256) {
    int rr = idx >> 8, j = idx & 255;
    Tf16[(size_t)(r0 + rr) * NUM_MODES + j] = (_Float16)Er[rr][j];
  }
}

// ---------------------------------------------------------------------------
// Phase 2: out[131072 x 64] = x[131072 x 256] @ T_eff^T.
// Workgroup = 256 threads = 8 waves; computes a 32x64 output block as
// 2 (M) x 4 (N) WMMA tiles of 16x16, K-loop of 8 x v_wmma_f32_16x16x32_f16.
// ---------------------------------------------------------------------------
__global__ __launch_bounds__(256) void wmma_gemm(
    const float* __restrict__ x, const _Float16* __restrict__ Tf16,
    float* __restrict__ out)
{
  __shared__ __attribute__((aligned(32))) _Float16 sA[32][IN_FEAT];        // 16 KB
  __shared__ __attribute__((aligned(32))) _Float16 sB[OUT_FEAT][IN_FEAT];  // 32 KB

  const int t = threadIdx.x;
  const size_t rowBase = (size_t)blockIdx.x * 32;

  // Stage B = T_eff f16 (L2-hot, 32 KB) into LDS with async DMA:
  // 2048 x 16B chunks, 8 per thread. Per lane: LDS[VDST] <- MEM[VADDR],
  // tracked by ASYNCcnt; overlaps with the x fp32->f16 staging below.
  {
    unsigned ldsB = (unsigned)(unsigned long long)
        (__attribute__((address_space(3))) char*)&sB[0][0];
    const char* gB = (const char*)Tf16;
#pragma unroll
    for (int i = 0; i < 8; ++i) {
      const int chunk = t + i * 256;
      unsigned           lAddr = ldsB + chunk * 16;
      unsigned long long gAddr = (unsigned long long)(gB + (size_t)chunk * 16);
      asm volatile("global_load_async_to_lds_b128 %0, %1, off"
                   :: "v"(lAddr), "v"(gAddr) : "memory");
    }
  }

  // Stage A: 32 rows of x, fp32 -> f16 on the way into LDS.
  // thread t: row = t>>3, 32-float segment = t&7. Nontemporal (streamed once).
  {
    const int row = t >> 3, seg = t & 7;
    const v4f* gx = (const v4f*)(x + (rowBase + row) * IN_FEAT + seg * 32);
#pragma unroll
    for (int i = 0; i < 8; ++i) {
      v4f f = __builtin_nontemporal_load(&gx[i]);
      v4h h = { (_Float16)f[0], (_Float16)f[1], (_Float16)f[2], (_Float16)f[3] };
      *(v4h*)&sA[row][seg * 32 + i * 4] = h;
    }
  }

  // Each wave drains its own async copies, then the barrier makes the whole
  // sB tile visible workgroup-wide.
  asm volatile("s_wait_asynccnt 0x0" ::: "memory");
  __syncthreads();

  const int wave = t >> 5;
  const int lane = t & 31;
  const int h    = lane >> 4;        // lane half
  const int mi   = lane & 15;        // M (for A) / N (for B) within tile
  const int m0   = (wave >> 2) * 16; // tile row base in sA
  const int n0   = (wave & 3) * 16;  // tile col base (row in sB)

  v8f acc = {};
#pragma unroll
  for (int k0 = 0; k0 < IN_FEAT; k0 += 32) {
    // A fragment: elems 0..15 <-> K = k0+16h+i contiguous (matches B)
    //
    v8h alo = *(const v8h*)&sA[m0 + mi][k0 + h * 16];
    v8h ahi = *(const v8h*)&sA[m0 + mi][k0 + h * 16 + 8];
    v16h a = __builtin_shufflevector(alo, ahi,
                                     0, 1, 2, 3, 4, 5, 6, 7,
                                     8, 9, 10, 11, 12, 13, 14, 15);
    // B fragment: lane (h, n=mi) holds K = k0+16h .. k0+16h+15 contiguous
    v16h b = *(const v16h*)&sB[n0 + mi][k0 + h * 16];

    acc = __builtin_amdgcn_wmma_f32_16x16x32_f16(
        /*neg_a=*/false, a, /*neg_b=*/false, b,
        /*c_mod=*/(short)0, acc, /*reuse_a=*/false, /*reuse_b=*/false);
  }

  // C/D layout: VGPR r, lanes 0-15 -> (M=r, N=lane), lanes 16-31 -> (M=r+8).
  float* o = out + (rowBase + m0 + h * 8) * OUT_FEAT + n0 + mi;
#pragma unroll
  for (int r = 0; r < 8; ++r)
    __builtin_nontemporal_store(acc[r], o + (size_t)r * OUT_FEAT);
}

// ---------------------------------------------------------------------------
extern "C" void kernel_launch(void* const* d_in, const int* in_sizes, int n_in,
                              void* d_out, int out_size, void* d_ws, size_t ws_size,
                              hipStream_t stream) {
  const float* x      = (const float*)d_in[0];
  const float* theta  = (const float*)d_in[1];
  const float* phi    = (const float*)d_in[2];
  const float* opha   = (const float*)d_in[3];
  const int*   mode1  = (const int*)d_in[4];
  const int*   mode2  = (const int*)d_in[5];
  float*       out    = (float*)d_out;
  _Float16*    Tf16   = (_Float16*)d_ws;   // 64*256*2 = 32 KB scratch

  build_T<<<dim3(16), dim3(256), 0, stream>>>(theta, phi, opha, mode1, mode2, Tf16);

  const int batch = in_sizes[0] / IN_FEAT;     // 131072
  wmma_gemm<<<dim3(batch / 32), dim3(256), 0, stream>>>(x, Tf16, out);
}
